// Attention_71064528879799
// MI455X (gfx1250) — matrix operand order repaired and checked by reading, not compiled, not run on hardware
//
#include <hip/hip_runtime.h>
#include <hip/hip_bf16.h>

// ---------------------------------------------------------------------------
// CDNA5 (gfx1250, wave32) attention forward, fused-softmax formulation:
//   Stage 1: q = (x@Wq)*scale ; k = x@Wk ; v = x@Wv      (WMMA GEMM)
//   Stage 2: fused per (b,h, 64-q-row block):
//       pass 1: online row max/sum of  S = q@k^T + bias
//       pass 2: recompute S, write attn = exp(S-m)/s ONCE, accumulate H = P@v
//   Stage 3: out = H@Wproj                               (WMMA GEMM)
// attn HBM traffic: single 402 MB write (vs 4 passes unfused). Per-head K/V
// panels (256 KB) stay L2-resident (192 MB L2) across the 16 q-row blocks.
// All matmuls: v_wmma_f32_16x16x32_f16. LDS tiles in WMMA fragment order
// (contiguous 32 B per lane per 16x16 subtile -> ds_load_b128 gathers).
// ---------------------------------------------------------------------------

typedef __attribute__((ext_vector_type(16))) _Float16 v16h;
typedef __attribute__((ext_vector_type(8)))  float    v8f;

static __device__ __forceinline__ unsigned pack2h(float a, float b) {
    union { _Float16 h[2]; unsigned u; } u_;
    u_.h[0] = (_Float16)a;
    u_.h[1] = (_Float16)b;
    return u_.u;
}

// ============================ generic WMMA GEMM ============================
// C = alpha * A @ B   (f32 in/out, f16 compute). M%64==0, N%64==0, K%32==0.
#define TILE_M 64
#define TILE_N 64
#define TILE_K 32

__global__ __launch_bounds__(128)
void wmma_gemm_kernel(const float* __restrict__ A, long long lda,
                      const float* __restrict__ B, long long ldb,
                      float* __restrict__ C, long long ldc,
                      int M, int N, int K, float alpha)
{
    __shared__ v16h AsF[4 * 32];   // 4 KB : A 64x32 tile, fragment order
    __shared__ v16h BsF[4 * 32];   // 4 KB : B 32x64 tile, fragment order
    _Float16* __restrict__ BsH  = (_Float16*)BsF;
    unsigned* __restrict__ As32 = (unsigned*)AsF;

    const int tid  = threadIdx.x;
    const int lane = tid & 31;
    const int wave = tid >> 5;

    const long long row0 = (long long)blockIdx.x * TILE_M;
    const long long col0 = (long long)blockIdx.y * TILE_N;

    v8f acc[4];
#pragma unroll
    for (int j = 0; j < 4; ++j) acc[j] = (v8f){0.f,0.f,0.f,0.f,0.f,0.f,0.f,0.f};

    for (int k0 = 0; k0 < K; k0 += TILE_K) {
        // A 64x32: pairs over k -> packed b32 stores in fragment order
#pragma unroll 4
        for (int i = tid; i < TILE_M * (TILE_K / 2); i += 128) {
            const int r  = i >> 4;
            const int cp = (i & 15) << 1;
            const float a0 = A[(row0 + r) * lda + (long long)(k0 + cp)];
            const float a1 = A[(row0 + r) * lda + (long long)(k0 + cp + 1)];
            const int lane_ = (r & 15) + ((cp >> 4) << 4);
            const int e     = cp & 15;
            As32[(((r >> 4) * 32 + lane_) * 16 + e) >> 1] = pack2h(a0, a1);
        }
        // B 32x64 row-major: pairs over n (coalesced global), two b16 stores
#pragma unroll 4
        for (int i = tid; i < TILE_K * (TILE_N / 2); i += 128) {
            const int kk = i >> 5;
            const int np = (i & 31) << 1;
            const float b0 = B[(long long)(k0 + kk) * ldb + (col0 + np)];
            const float b1 = B[(long long)(k0 + kk) * ldb + (col0 + np + 1)];
            const int lgrp = (kk >> 4) << 4;
            const int e    = kk & 15;
            BsH[((np >> 4) * 32 + ((np & 15) + lgrp)) * 16 + e]       = (_Float16)b0;
            BsH[((np >> 4) * 32 + (((np + 1) & 15) + lgrp)) * 16 + e] = (_Float16)b1;
        }
        __syncthreads();

        const v16h afrag = AsF[wave * 32 + lane];
#pragma unroll
        for (int j = 0; j < 4; ++j) {
            const v16h bfrag = BsF[j * 32 + lane];
            acc[j] = __builtin_amdgcn_wmma_f32_16x16x32_f16(
                         false, afrag, false, bfrag, (short)0, acc[j], false, false);
        }
        __syncthreads();
    }

    const int g  = lane >> 4;
    const int nl = lane & 15;
#pragma unroll
    for (int j = 0; j < 4; ++j) {
#pragma unroll
        for (int r = 0; r < 8; ++r) {
            const long long gm = row0 + wave * 16 + r + (g << 3);
            const long long gn = col0 + (j << 4) + nl;
            C[gm * ldc + gn] = alpha * acc[j][r];
        }
    }
}

// ===================== fused attention core (stages 2-4) ====================
// grid: (16 q-row-blocks, 96 z=b*12+h), block 128 (4 waves).
// Q/K/V: [8192 x 768] f32 (head h at cols h*64..h*64+63); bias [1024x1024];
// attn out: [96 x 1024 x 1024]; H out: [8192 x 768].
#define LDC 768LL
#define LDA_ATTN 1024LL
#define SS_STR 65

__global__ __launch_bounds__(128)
void fused_attn_kernel(const float* __restrict__ Q,
                       const float* __restrict__ Km,
                       const float* __restrict__ Vm,
                       const float* __restrict__ bias,
                       float* __restrict__ attn,
                       float* __restrict__ H)
{
    __shared__ v16h QsF[2 * 128];     // 8 KB  A-frag: Q block 64x64 (2 ksteps)
    __shared__ v16h KsF[2 * 128];     // 8 KB  B-frag: K^T tile  64x64
    __shared__ v16h VsF[2 * 128];     // 8 KB  B-frag: V tile    64x64
    __shared__ v16h PsF[2 * 128];     // 8 KB  A-frag: P tile    64x64
    __shared__ float Ss[64 * SS_STR]; // 16.25 KB  S tile f32 (padded)
    __shared__ float rowM[64];
    __shared__ float rowS[64];

    const int tid  = threadIdx.x;
    const int lane = tid & 31;
    const int wave = tid >> 5;
    const int g    = lane >> 4;
    const int nl   = lane & 15;

    const int z  = blockIdx.y;
    const long long bb = z / 12;
    const long long hh = z % 12;
    const long long q0 = (long long)blockIdx.x * 64;

    const float* __restrict__ Qp = Q  + bb * 1024 * LDC + hh * 64;
    const float* __restrict__ Kp = Km + bb * 1024 * LDC + hh * 64;
    const float* __restrict__ Vp = Vm + bb * 1024 * LDC + hh * 64;
    float* __restrict__ attnP = attn + (long long)z * (1024LL * 1024LL);
    float* __restrict__ Hp    = H  + bb * 1024 * LDC + hh * 64;

    unsigned* __restrict__ Qs32 = (unsigned*)QsF;
    unsigned* __restrict__ Ks32 = (unsigned*)KsF;
    unsigned* __restrict__ Ps32 = (unsigned*)PsF;
    _Float16* __restrict__ VsH  = (_Float16*)VsF;

    // ---- stage Q block once: 64 rows x 64 d, A-fragment order --------------
#pragma unroll 4
    for (int i = tid; i < 64 * 32; i += 128) {     // pairs over k
        const int r  = i >> 5;
        const int kp = (i & 31) << 1;
        const float a0 = Qp[(q0 + r) * LDC + kp];
        const float a1 = Qp[(q0 + r) * LDC + kp + 1];
        const int ks = kp >> 5, k2 = kp & 31;
        const int lane_ = (r & 15) + ((k2 >> 4) << 4);
        const int e     = k2 & 15;
        Qs32[((ks * 128 + (r >> 4) * 32 + lane_) * 16 + e) >> 1] = pack2h(a0, a1);
    }
    if (tid < 64) { rowM[tid] = -3.402823466e38f; rowS[tid] = 0.f; }

    v8f hacc[4];
#pragma unroll
    for (int j = 0; j < 4; ++j) hacc[j] = (v8f){0.f,0.f,0.f,0.f,0.f,0.f,0.f,0.f};
    __syncthreads();

    // =========================== PASS 1 + PASS 2 ===========================
#pragma unroll 1
    for (int pass = 0; pass < 2; ++pass) {
#pragma unroll 1
        for (int t = 0; t < 16; ++t) {
            // ---- stage K tile (keys t*64.. +63) in B-frag order -----------
            // B[k=d][n=key]: global read K[(key)*768 + d] -> pairs over d
#pragma unroll 4
            for (int i = tid; i < 64 * 32; i += 128) {
                const int n  = i >> 5;
                const int kp = (i & 31) << 1;
                const float b0 = Kp[((long long)t * 64 + n) * LDC + kp];
                const float b1 = Kp[((long long)t * 64 + n) * LDC + kp + 1];
                const int ks = kp >> 5, k2 = kp & 31;
                const int lane_ = (n & 15) + ((k2 >> 4) << 4);
                Ks32[((ks * 128 + (n >> 4) * 32 + lane_) * 16 + (k2 & 15)) >> 1]
                    = pack2h(b0, b1);
            }
            __syncthreads();

            // ---- S tile = Q @ K^T (two k-steps of 32) ---------------------
            v8f sacc[4];
#pragma unroll
            for (int j = 0; j < 4; ++j) sacc[j] = (v8f){0.f,0.f,0.f,0.f,0.f,0.f,0.f,0.f};
#pragma unroll
            for (int ks = 0; ks < 2; ++ks) {
                const v16h a = QsF[ks * 128 + wave * 32 + lane];
#pragma unroll
                for (int j = 0; j < 4; ++j) {
                    const v16h bf = KsF[ks * 128 + j * 32 + lane];
                    sacc[j] = __builtin_amdgcn_wmma_f32_16x16x32_f16(
                                  false, a, false, bf, (short)0, sacc[j], false, false);
                }
            }
            // store S tile (+bias) to LDS
#pragma unroll
            for (int j = 0; j < 4; ++j) {
#pragma unroll
                for (int r = 0; r < 8; ++r) {
                    const int row = wave * 16 + r + (g << 3);
                    const int col = (j << 4) + nl;
                    Ss[row * SS_STR + col] =
                        sacc[j][r] + bias[(q0 + row) * LDA_ATTN + (long long)t * 64 + col];
                }
            }
            __syncthreads();

            if (pass == 0) {
                // ---- online row max/sum update (one row per thread<64) ----
                if (tid < 64) {
                    const int row = tid;
                    float mT = -3.402823466e38f;
                    for (int c = 0; c < 64; ++c)
                        mT = fmaxf(mT, Ss[row * SS_STR + c]);
                    const float mOld = rowM[row];
                    const float mNew = fmaxf(mOld, mT);
                    float acc = rowS[row] * __expf(mOld - mNew);
                    for (int c = 0; c < 64; ++c)
                        acc += __expf(Ss[row * SS_STR + c] - mNew);
                    rowM[row] = mNew;
                    rowS[row] = acc;
                }
                __syncthreads();
            } else {
                // ---- stage V tile in B-frag order (pairs over d cols) -----
#pragma unroll 4
                for (int i = tid; i < 64 * 32; i += 128) {
                    const int kk = i >> 5;            // key within tile
                    const int np = (i & 31) << 1;     // even d col
                    const float v0 = Vp[((long long)t * 64 + kk) * LDC + np];
                    const float v1 = Vp[((long long)t * 64 + kk) * LDC + np + 1];
                    const int ks = kk >> 5, k2 = kk & 31;
                    const int lgrp = (k2 >> 4) << 4;
                    const int e    = k2 & 15;
                    VsH[(ks * 128 + (np >> 4) * 32 + ((np & 15) + lgrp)) * 16 + e]
                        = (_Float16)v0;
                    VsH[(ks * 128 + (np >> 4) * 32 + (((np + 1) & 15) + lgrp)) * 16 + e]
                        = (_Float16)v1;
                }
                // ---- P = exp(S-m)/s : write attn + stage A-frag P ---------
                {
                    const int row = tid >> 1;
                    const int c0  = (tid & 1) << 5;       // 0 or 32 == kstep
                    const float mR   = rowM[row];
                    const float invS = 1.f / rowS[row];
                    float* __restrict__ aRow =
                        attnP + (q0 + row) * LDA_ATTN + (long long)t * 64 + c0;
                    const int ks  = c0 >> 5;
                    const int sub = row >> 4;
                    const int m   = row & 15;
#pragma unroll 4
                    for (int cc = 0; cc < 32; cc += 2) {
                        const float p0 = __expf(Ss[row * SS_STR + c0 + cc]     - mR) * invS;
                        const float p1 = __expf(Ss[row * SS_STR + c0 + cc + 1] - mR) * invS;
                        aRow[cc]     = p0;
                        aRow[cc + 1] = p1;
                        const int lane_ = m + ((cc >> 4) << 4);
                        const int e     = cc & 15;
                        Ps32[((ks * 128 + sub * 32 + lane_) * 16 + e) >> 1] = pack2h(p0, p1);
                    }
                }
                __syncthreads();

                // ---- H += P @ V ------------------------------------------
#pragma unroll
                for (int ks = 0; ks < 2; ++ks) {
                    const v16h a = PsF[ks * 128 + wave * 32 + lane];
#pragma unroll
                    for (int j = 0; j < 4; ++j) {
                        const v16h bf = VsF[ks * 128 + j * 32 + lane];
                        hacc[j] = __builtin_amdgcn_wmma_f32_16x16x32_f16(
                                      false, a, false, bf, (short)0, hacc[j], false, false);
                    }
                }
                __syncthreads();
            }
        }
    }

    // ---- store H block ------------------------------------------------------
#pragma unroll
    for (int j = 0; j < 4; ++j) {
#pragma unroll
        for (int r = 0; r < 8; ++r) {
            const int row = wave * 16 + r + (g << 3);
            const int col = (j << 4) + nl;
            Hp[(q0 + row) * LDC + col] = hacc[j][r];
        }
    }
}

// ================================ launcher =================================
extern "C" void kernel_launch(void* const* d_in, const int* in_sizes, int n_in,
                              void* d_out, int out_size, void* d_ws, size_t ws_size,
                              hipStream_t stream) {
    (void)in_sizes; (void)n_in; (void)out_size; (void)ws_size;

    const float* x     = (const float*)d_in[0];   // [8,1024,768]
    const float* Wq    = (const float*)d_in[1];   // [768,768]
    const float* Wk    = (const float*)d_in[2];
    const float* Wv    = (const float*)d_in[3];
    const float* Wproj = (const float*)d_in[4];
    const float* bias  = (const float*)d_in[5];   // [1024,1024]

    const long long Bn = 8, N = 1024, C = 768;
    const long long M  = Bn * N;                  // 8192
    const float scale  = 0.125f;                  // 64^-0.5

    float* out  = (float*)d_out;                  // [8,1024,768]
    float* attn = out + M * C;                    // [8,12,1024,1024]

    const long long QN = M * C;                   // 25.2 MB each
    float* ws = (float*)d_ws;
    float* Q  = ws;
    float* Kp = ws + QN;
    float* V  = ws + 2 * QN;
    float* H  = ws + 3 * QN;

    const dim3 blk(128);

    // Stage 1: Q/K/V projections  [8192x768] @ [768x768]
    wmma_gemm_kernel<<<dim3(M/64, C/64), blk, 0, stream>>>(
        x, C, Wq, C, Q,  C, (int)M, (int)C, (int)C, scale);
    wmma_gemm_kernel<<<dim3(M/64, C/64), blk, 0, stream>>>(
        x, C, Wk, C, Kp, C, (int)M, (int)C, (int)C, 1.0f);
    wmma_gemm_kernel<<<dim3(M/64, C/64), blk, 0, stream>>>(
        x, C, Wv, C, V,  C, (int)M, (int)C, (int)C, 1.0f);

    // Stage 2: fused scores+bias+softmax+attn-write+P@V
    fused_attn_kernel<<<dim3(16, 96), blk, 0, stream>>>(Q, Kp, V, bias, attn, H);

    // Stage 3: out = H @ Wproj
    wmma_gemm_kernel<<<dim3(M/64, C/64), blk, 0, stream>>>(
        H, C, Wproj, C, out, C, (int)M, (int)C, (int)C, 1.0f);
}
